// NystromAttention_29781303230530
// MI455X (gfx1250) — compile-verified, matrix operand order checked by reading, not run on hardware
//
#include <hip/hip_runtime.h>
#include <stdint.h>

typedef __bf16 bf16;
typedef __bf16 v16bf __attribute__((ext_vector_type(16)));
typedef __bf16 v8bf  __attribute__((ext_vector_type(8)));
typedef float  v8f   __attribute__((ext_vector_type(8)));
typedef int    vsi4  __attribute__((__vector_size__(16)));   // 4 x int, b128 payload

#define NB   4
#define NSEQ 8192
#define NF   512
#define NH   8
#define NM   256
#define ND   64
#define NKS  33
#define QSCALE 0.125f      // 64^-0.5
#define LPM  32            // tokens per landmark
#define BH   (NB*NH)       // 32

#if defined(__has_builtin)
#if __has_builtin(__builtin_amdgcn_global_load_async_to_lds_b128) && \
    __has_builtin(__builtin_amdgcn_s_wait_asynccnt)
#define HAVE_ASYNC_LDS 1
#endif
#endif
#ifndef HAVE_ASYNC_LDS
#define HAVE_ASYNC_LDS 0
#endif

// ---------------------------------------------------------------------------
// Fragment load: 16 bf16 per lane, NT layout (row-major, K contiguous).
// lanes 0-15 (half=0): K = kk+0..7, kk+16..23 ; lanes 16-31: K = kk+8..15, kk+24..31
// ---------------------------------------------------------------------------
__device__ __forceinline__ v16bf load_frag_nt(const bf16* __restrict__ rowptr,
                                              int kk, int half) {
  const bf16* p = rowptr + kk + half * 8;
  v8bf lo = *(const v8bf*)(p);
  v8bf hi = *(const v8bf*)(p + 16);
  v16bf f;
#pragma unroll
  for (int i = 0; i < 8; ++i) { f[i] = lo[i]; f[i + 8] = hi[i]; }
  return f;
}

// ---------------------------------------------------------------------------
// NT GEMM: C[M,N] = alpha * A[M,K] * B[N,K]^T  (+ bias[col])
// Wave tile: 64 (M) x 32 (N) -> 4x2 grid of 16x16 WMMA tiles, 8 WMMAs/K-step.
// OUT_MODE: 0 = no normal output, 1 = f32 normal, 2 = bf16 normal
// TRANS_OUT: also write bf16 C^T (Ct[col*ldct + row])
// ---------------------------------------------------------------------------
template <int OUT_MODE, bool TRANS_OUT, bool BIAS>
__global__ __launch_bounds__(256) void wmma_gemm_nt(
    const bf16* __restrict__ A, const bf16* __restrict__ Bm,
    float* __restrict__ Cf, bf16* __restrict__ Cb, bf16* __restrict__ Ct,
    const float* __restrict__ bias,
    int M, int N, int K,
    long long sA, long long sB, long long sC, long long sCt,
    int ldc, int ldct, float alpha, int tM, int tN) {
  int batch = blockIdx.y;
  A  += sA * batch;
  Bm += sB * batch;
  int wid  = threadIdx.x >> 5;
  int tile = blockIdx.x * 8 + wid;
  if (tile >= tM * tN) return;           // wave-uniform exit: EXEC stays all-1s
  int tm   = tile % tM;                  // M-tile (64 rows); fastest -> B reuse
  int tn   = tile / tM;                  // N-tile (32 cols)
  int lane = threadIdx.x & 31;
  int half = lane >> 4;
  int r    = lane & 15;

  const bf16* ar[4];
  const bf16* br[2];
#pragma unroll
  for (int i = 0; i < 4; ++i) ar[i] = A  + (long long)(tm * 64 + i * 16 + r) * K;
#pragma unroll
  for (int j = 0; j < 2; ++j) br[j] = Bm + (long long)(tn * 32 + j * 16 + r) * K;

  v8f acc[4][2] = {};
  for (int kk = 0; kk < K; kk += 32) {
    if (kk + 128 < K) {
#pragma unroll
      for (int i = 0; i < 4; ++i) __builtin_prefetch(ar[i] + kk + 128, 0, 0);
#pragma unroll
      for (int j = 0; j < 2; ++j) __builtin_prefetch(br[j] + kk + 128, 0, 0);
    }
    v16bf af[4], bfr[2];
#pragma unroll
    for (int i = 0; i < 4; ++i) af[i] = load_frag_nt(ar[i], kk, half);
#pragma unroll
    for (int j = 0; j < 2; ++j) bfr[j] = load_frag_nt(br[j], kk, half);
#pragma unroll
    for (int i = 0; i < 4; ++i)
#pragma unroll
      for (int j = 0; j < 2; ++j)
        acc[i][j] = __builtin_amdgcn_wmma_f32_16x16x32_bf16(
            false, af[i], false, bfr[j], (short)0, acc[i][j], false, false);
  }

#pragma unroll
  for (int j = 0; j < 2; ++j) {
    int col  = tn * 32 + j * 16 + r;
    float bv = BIAS ? bias[col] : 0.0f;
#pragma unroll
    for (int i = 0; i < 4; ++i) {
#pragma unroll
      for (int e = 0; e < 8; ++e) {
        int row = tm * 64 + i * 16 + half * 8 + e;
        float v = acc[i][j][e] * alpha + bv;
        if (OUT_MODE == 1) Cf[sC * batch + (long long)row * ldc + col] = v;
        if (OUT_MODE == 2) Cb[sC * batch + (long long)row * ldc + col] = (bf16)v;
        if (TRANS_OUT)     Ct[sCt * batch + (long long)col * ldct + row] = (bf16)v;
      }
    }
  }
}

// ---------------------------------------------------------------------------
// Elementwise helpers
// ---------------------------------------------------------------------------
__global__ void cvt_f32_bf16(const float* __restrict__ in, bf16* __restrict__ out,
                             long long n) {
  long long i = (long long)blockIdx.x * 256 + threadIdx.x;
  if (i < n) out[i] = (bf16)in[i];
}

// Split qkv f32 (token-major, 1536 feats) into head-major bf16 q (scaled), k, v, v^T
__global__ void pack_qkv(const float* __restrict__ qkv, bf16* __restrict__ q,
                         bf16* __restrict__ k, bf16* __restrict__ v,
                         bf16* __restrict__ vT) {
  long long i = (long long)blockIdx.x * 256 + threadIdx.x;
  if (i >= (long long)NB * NSEQ * 3 * NF) return;
  int f        = (int)(i % (3 * NF));
  long long tok = i / (3 * NF);            // b*NSEQ + t
  int which = f >> 9;                       // 0:q 1:k 2:v
  int fr    = f & (NF - 1);
  int h     = fr >> 6;
  int d     = fr & (ND - 1);
  long long b  = tok >> 13;
  long long t  = tok & (NSEQ - 1);
  long long bh = b * NH + h;
  float val = qkv[i];
  long long di = (bh * NSEQ + t) * ND + d;
  if (which == 0)      q[di] = (bf16)(val * QSCALE);
  else if (which == 1) k[di] = (bf16)val;
  else {
    v[di] = (bf16)val;
    vT[(bh * ND + d) * NSEQ + t] = (bf16)val;
  }
}

// landmark means over groups of 32 tokens: (BH, 256, 64)
__global__ void landmarks(const bf16* __restrict__ src, bf16* __restrict__ dst) {
  long long i = (long long)blockIdx.x * 256 + threadIdx.x;
  if (i >= (long long)BH * NM * ND) return;
  int d  = (int)(i & (ND - 1));
  int m  = (int)((i >> 6) & (NM - 1));
  int bh = (int)(i >> 14);
  float s = 0.f;
#pragma unroll 4
  for (int j = 0; j < LPM; ++j)
    s += (float)src[((long long)bh * NSEQ + m * LPM + j) * ND + d];
  dst[i] = (bf16)(s * (1.f / LPM));
}

// row softmax: f32 in -> bf16 out, one block per row
__global__ void softmax_rows(const float* __restrict__ in, bf16* __restrict__ out,
                             int cols) {
  long long row = blockIdx.x;
  const float* rp = in + row * (long long)cols;
  bf16* op = out + row * (long long)cols;
  __shared__ float red[256];
  float m = -1e30f;
  for (int j = threadIdx.x; j < cols; j += 256) m = fmaxf(m, rp[j]);
  red[threadIdx.x] = m; __syncthreads();
  for (int s = 128; s > 0; s >>= 1) {
    if (threadIdx.x < s) red[threadIdx.x] = fmaxf(red[threadIdx.x], red[threadIdx.x + s]);
    __syncthreads();
  }
  m = red[0]; __syncthreads();
  float sum = 0.f;
  for (int j = threadIdx.x; j < cols; j += 256) sum += __expf(rp[j] - m);
  red[threadIdx.x] = sum; __syncthreads();
  for (int s = 128; s > 0; s >>= 1) {
    if (threadIdx.x < s) red[threadIdx.x] += red[threadIdx.x + s];
    __syncthreads();
  }
  float inv = 1.f / red[0];
  for (int j = threadIdx.x; j < cols; j += 256)
    op[j] = (bf16)(__expf(rp[j] - m) * inv);
}

__global__ void init_stats(float* s) { if (threadIdx.x < 2) s[threadIdx.x] = 0.f; }

// global max of abs-sums along one axis of batched 256x256 bf16 matrices
__global__ void abs_sum_max(const bf16* __restrict__ x, float* __restrict__ outmax,
                            int s1, int s2) {
  int bh = blockIdx.x >> 8;
  int i  = blockIdx.x & 255;
  long long base = (long long)bh * NM * NM + (long long)i * s1;
  __shared__ float red[256];
  float sum = 0.f;
  for (int j = threadIdx.x; j < NM; j += 256)
    sum += fabsf((float)x[base + (long long)j * s2]);
  red[threadIdx.x] = sum; __syncthreads();
  for (int s = 128; s > 0; s >>= 1) {
    if (threadIdx.x < s) red[threadIdx.x] += red[threadIdx.x + s];
    __syncthreads();
  }
  if (threadIdx.x == 0) atomicMax((int*)outmax, __float_as_int(red[0]));
}

// z0 = x^T / c, z0^T = x / c   (c = stats[0]*stats[1])
__global__ void pinv_init(const bf16* __restrict__ x2, bf16* __restrict__ z,
                          bf16* __restrict__ zT, const float* __restrict__ stats) {
  long long idx = (long long)blockIdx.x * 256 + threadIdx.x;
  if (idx >= (long long)BH * NM * NM) return;
  float inv = 1.f / (stats[0] * stats[1]);
  int j  = (int)(idx & 255);
  int i  = (int)((idx >> 8) & 255);
  long long bh = idx >> 16;
  long long base = bh << 16;
  z[idx]  = (bf16)((float)x2[base + (long long)j * NM + i] * inv);
  zT[idx] = (bf16)((float)x2[idx] * inv);
}

// out = a*I - in  (batched 256x256, bf16). (aI - S)^T = aI - S^T, so works on transposes.
__global__ void aI_minus(const bf16* __restrict__ in, bf16* __restrict__ out, float a) {
  long long idx = (long long)blockIdx.x * 256 + threadIdx.x;
  if (idx >= (long long)BH * NM * NM) return;
  int j = (int)(idx & 255);
  int i = (int)((idx >> 8) & 255);
  out[idx] = (bf16)(((i == j) ? a : 0.f) - (float)in[idx]);
}

// ---------------------------------------------------------------------------
// Depthwise conv-33 residual, LDS-staged with async global->LDS copies.
// Block: one (bh, 64-token) tile; stages 96 tokens x 64 lanes (12 KB) into LDS.
// outh[bh,t,d] += sum_k wres[h,k] * v[bh,t+k-16,d]
// ---------------------------------------------------------------------------
#define CT 64  // tokens per block
__global__ __launch_bounds__(256) void conv_residual(const bf16* __restrict__ v,
                                                     const float* __restrict__ wres,
                                                     float* __restrict__ outh) {
  __shared__ bf16 tilebuf[(CT + 32) * ND];   // 96 x 64 bf16 = 12 KB
  int bh = blockIdx.y;
  int t0 = blockIdx.x * CT;
  int h  = bh & (NH - 1);
  const bf16* src = v + (long long)bh * NSEQ * ND;

  // stage rows t0-16 .. t0+CT+15 ; 16B chunks, 8 chunks per 64-wide row
  const int chunks = (CT + 32) * ND / 8;     // 768
  for (int c = threadIdx.x; c < chunks; c += 256) {
    int row = c >> 3;
    int col = (c & 7) * 8;
    long long t = (long long)t0 - 16 + row;
    bf16* lp = &tilebuf[row * ND + col];
    if (t >= 0 && t < NSEQ) {
#if HAVE_ASYNC_LDS
      __builtin_amdgcn_global_load_async_to_lds_b128(
          (__attribute__((address_space(1))) vsi4*)(void*)(src + t * ND + col),
          (__attribute__((address_space(3))) vsi4*)(void*)lp, 0, 0);
#else
      *(v8bf*)lp = *(const v8bf*)(src + t * ND + col);
#endif
    } else {
#pragma unroll
      for (int u = 0; u < 8; ++u) lp[u] = (bf16)0.f;
    }
  }
#if HAVE_ASYNC_LDS
  __builtin_amdgcn_s_wait_asynccnt(0);
#endif
  __syncthreads();

  int d = threadIdx.x & (ND - 1);
  for (int tt = threadIdx.x >> 6; tt < CT; tt += 4) {
    float acc = 0.f;
#pragma unroll
    for (int k = 0; k < NKS; ++k)
      acc += wres[h * NKS + k] * (float)tilebuf[(tt + k) * ND + d];
    outh[((long long)bh * NSEQ + t0 + tt) * ND + d] += acc;
  }
}

// (B,H,n,D) f32 -> (B,n,H*D) bf16
__global__ void merge_heads(const float* __restrict__ outh, bf16* __restrict__ y) {
  long long idx = (long long)blockIdx.x * 256 + threadIdx.x;
  if (idx >= (long long)NB * NSEQ * NF) return;
  int fd = (int)(idx & (NF - 1));
  int h  = fd >> 6;
  int d  = fd & (ND - 1);
  long long tok = idx >> 9;
  long long b = tok >> 13;
  long long t = tok & (NSEQ - 1);
  y[idx] = (bf16)outh[(((b * NH + h) * NSEQ) + t) * ND + d];
}

// ---------------------------------------------------------------------------
// Host-side launch helpers
// ---------------------------------------------------------------------------
template <int OM, bool TO, bool BIAS>
static inline void gemm_nt(hipStream_t s, const bf16* A, const bf16* B, float* Cf,
                           bf16* Cb, bf16* Ct, const float* bias, int M, int N,
                           int K, long long sA, long long sB, long long sC,
                           long long sCt, int ldc, int ldct, float alpha,
                           int batch) {
  int tM = M / 64, tN = N / 32;
  int T = tM * tN;
  dim3 grid((T + 7) / 8, batch, 1);
  wmma_gemm_nt<OM, TO, BIAS><<<grid, 256, 0, s>>>(A, B, Cf, Cb, Ct, bias, M, N, K,
                                                  sA, sB, sC, sCt, ldc, ldct,
                                                  alpha, tM, tN);
}

static inline int gblk(long long n) { return (int)((n + 255) / 256); }

extern "C" void kernel_launch(void* const* d_in, const int* in_sizes, int n_in,
                              void* d_out, int out_size, void* d_ws, size_t ws_size,
                              hipStream_t stream) {
  (void)in_sizes; (void)n_in; (void)out_size; (void)ws_size;
  const float* x_f32   = (const float*)d_in[0];
  const float* wqkv_f  = (const float*)d_in[1];
  const float* wout_f  = (const float*)d_in[2];
  const float* bout_f  = (const float*)d_in[3];
  const float* wres_f  = (const float*)d_in[4];
  float* outp          = (float*)d_out;

  // workspace carve-up
  char* w = (char*)d_ws;
  auto alloc = [&](size_t bytes) -> void* {
    void* p = (void*)w;
    w += (bytes + 255) & ~(size_t)255;
    return p;
  };
  const long long TOK = (long long)NB * NSEQ;           // 32768
  float* stats    = (float*)alloc(256);
  float* sim_f32  = (float*)alloc((long long)BH * NSEQ * NM * 4);   // 268MB (also qkv)
  float* qkv_f32  = sim_f32;                                        // alias: dead before sims
  bf16* x_bf      = (bf16*)alloc(TOK * NF * 2);
  bf16* wq_bf     = (bf16*)alloc((long long)3 * NF * NF * 2);
  bf16* wo_bf     = (bf16*)alloc((long long)NF * NF * 2);
  bf16* q_bf      = (bf16*)alloc(TOK * NF * 2);
  bf16* k_bf      = (bf16*)alloc(TOK * NF * 2);
  bf16* v_bf      = (bf16*)alloc(TOK * NF * 2);
  bf16* vT_bf     = (bf16*)alloc(TOK * NF * 2);
  bf16* q_land    = (bf16*)alloc((long long)BH * NM * ND * 2);
  bf16* k_land    = (bf16*)alloc((long long)BH * NM * ND * 2);
  bf16* attn1_bf  = (bf16*)alloc((long long)BH * NSEQ * NM * 2);
  bf16* attn3_bf  = (bf16*)alloc((long long)BH * NM * NSEQ * 2);
  float* attn2_f  = (float*)alloc((long long)BH * NM * NM * 4);
  bf16* x2_bf     = (bf16*)alloc((long long)BH * NM * NM * 2);
  bf16* z0        = (bf16*)alloc((long long)BH * NM * NM * 2);
  bf16* z1        = (bf16*)alloc((long long)BH * NM * NM * 2);
  bf16* zT0       = (bf16*)alloc((long long)BH * NM * NM * 2);
  bf16* zT1       = (bf16*)alloc((long long)BH * NM * NM * 2);
  bf16* xz_bf     = (bf16*)alloc((long long)BH * NM * NM * 2);
  bf16* xzT_bf    = (bf16*)alloc((long long)BH * NM * NM * 2);
  bf16* tbuf      = (bf16*)alloc((long long)BH * NM * NM * 2);
  bf16* sT_bf     = (bf16*)alloc((long long)BH * NM * NM * 2);
  bf16* a3vT      = (bf16*)alloc((long long)BH * ND * NM * 2);
  bf16* w2T       = (bf16*)alloc((long long)BH * ND * NM * 2);
  float* outh     = (float*)alloc(TOK * NF * 4);
  bf16* y_bf      = (bf16*)alloc(TOK * NF * 2);

  const long long sQ = (long long)NSEQ * ND;      // per-head q/k/v stride
  const long long sL = (long long)NM * ND;        // landmark stride
  const long long sMM = (long long)NM * NM;       // 256x256 stride

  // 1) convert inputs to bf16
  cvt_f32_bf16<<<gblk(TOK * NF), 256, 0, stream>>>(x_f32, x_bf, TOK * NF);
  cvt_f32_bf16<<<gblk(3LL * NF * NF), 256, 0, stream>>>(wqkv_f, wq_bf, 3LL * NF * NF);
  cvt_f32_bf16<<<gblk((long long)NF * NF), 256, 0, stream>>>(wout_f, wo_bf, (long long)NF * NF);

  // 2) qkv = x @ wqkv^T   (32768 x 1536, K=512)
  gemm_nt<1, false, false>(stream, x_bf, wq_bf, qkv_f32, nullptr, nullptr, nullptr,
                           (int)TOK, 3 * NF, NF, 0, 0, 0, 0, 3 * NF, 0, 1.f, 1);
  // 3) split into per-head q (scaled), k, v, v^T
  pack_qkv<<<gblk(TOK * 3 * NF), 256, 0, stream>>>(qkv_f32, q_bf, k_bf, v_bf, vT_bf);
  // 4) landmarks
  landmarks<<<gblk((long long)BH * NM * ND), 256, 0, stream>>>(q_bf, q_land);
  landmarks<<<gblk((long long)BH * NM * ND), 256, 0, stream>>>(k_bf, k_land);

  // 5) sim2 = q_land @ k_land^T -> softmax -> x2 (bf16)
  gemm_nt<1, false, false>(stream, q_land, k_land, attn2_f, nullptr, nullptr, nullptr,
                           NM, NM, ND, sL, sL, sMM, 0, NM, 0, 1.f, BH);
  softmax_rows<<<BH * NM, 256, 0, stream>>>(attn2_f, x2_bf, NM);

  // 6) sim1 = q @ k_land^T -> softmax -> attn1 (bf16)
  gemm_nt<1, false, false>(stream, q_bf, k_land, sim_f32, nullptr, nullptr, nullptr,
                           NSEQ, NM, ND, sQ, sL, (long long)NSEQ * NM, 0, NM, 0, 1.f, BH);
  softmax_rows<<<BH * NSEQ, 256, 0, stream>>>(sim_f32, attn1_bf, NM);

  // 7) sim3 = q_land @ k^T -> softmax -> attn3 (bf16)  (reuses sim buffer)
  gemm_nt<1, false, false>(stream, q_land, k_bf, sim_f32, nullptr, nullptr, nullptr,
                           NM, NSEQ, ND, sL, sQ, (long long)NM * NSEQ, 0, NSEQ, 0, 1.f, BH);
  softmax_rows<<<BH * NM, 256, 0, stream>>>(sim_f32, attn3_bf, NSEQ);

  // 8) pinv init: c = max(rowsum)*max(colsum) over all batches; z0 = x^T/c
  init_stats<<<1, 32, 0, stream>>>(stats);
  abs_sum_max<<<BH * NM, 256, 0, stream>>>(x2_bf, &stats[0], NM, 1);  // row sums
  abs_sum_max<<<BH * NM, 256, 0, stream>>>(x2_bf, &stats[1], 1, NM);  // col sums
  pinv_init<<<gblk((long long)BH * sMM), 256, 0, stream>>>(x2_bf, z0, zT0, stats);

  // 9) Newton-Schulz iterations, all-NT via transposed epilogues
  for (int it = 0; it < 6; ++it) {
    bf16* zc  = (it & 1) ? z1 : z0;
    bf16* zTc = (it & 1) ? zT1 : zT0;
    bf16* zn  = (it & 1) ? z0 : z1;
    bf16* zTn = (it & 1) ? zT0 : zT1;
    // xz = x @ z = NT(x, z^T); keep both xz and xz^T
    gemm_nt<2, true, false>(stream, x2_bf, zTc, nullptr, xz_bf, xzT_bf, nullptr,
                            NM, NM, NM, sMM, sMM, sMM, sMM, NM, NM, 1.f, BH);
    // t1^T = 7I - xz^T ; s1^T = (xz @ t1)^T
    aI_minus<<<gblk((long long)BH * sMM), 256, 0, stream>>>(xzT_bf, tbuf, 7.f);
    gemm_nt<0, true, false>(stream, xz_bf, tbuf, nullptr, nullptr, sT_bf, nullptr,
                            NM, NM, NM, sMM, sMM, 0, sMM, 0, NM, 1.f, BH);
    // t2^T = 15I - s1^T ; s2^T = (xz @ t2)^T
    aI_minus<<<gblk((long long)BH * sMM), 256, 0, stream>>>(sT_bf, tbuf, 15.f);
    gemm_nt<0, true, false>(stream, xz_bf, tbuf, nullptr, nullptr, sT_bf, nullptr,
                            NM, NM, NM, sMM, sMM, 0, sMM, 0, NM, 1.f, BH);
    // t3^T = 13I - s2^T ; z_new = 0.25 * z @ t3 (keep both z and z^T)
    aI_minus<<<gblk((long long)BH * sMM), 256, 0, stream>>>(sT_bf, tbuf, 13.f);
    gemm_nt<2, true, false>(stream, zc, tbuf, nullptr, zn, zTn, nullptr,
                            NM, NM, NM, sMM, sMM, sMM, sMM, NM, NM, 0.25f, BH);
  }
  // after 6 iters result sits in z0 / zT0

  // 10) attn3v^T = (attn3 @ v)^T   (NT: B = v^T, 64 x 8192)
  gemm_nt<0, true, false>(stream, attn3_bf, vT_bf, nullptr, nullptr, a3vT, nullptr,
                          NM, ND, NSEQ, (long long)NM * NSEQ, sQ, 0,
                          (long long)ND * NM, 0, NM, 1.f, BH);
  // 11) w2^T = (z_pinv @ attn3v)^T  (NT: B = attn3v^T, 64 x 256)
  gemm_nt<0, true, false>(stream, z0, a3vT, nullptr, nullptr, w2T, nullptr,
                          NM, ND, NM, sMM, (long long)ND * NM, 0,
                          (long long)ND * NM, 0, NM, 1.f, BH);
  // 12) out_h = attn1 @ w2  (NT: B = w2^T) -> f32 (B,H,n,D)
  gemm_nt<1, false, false>(stream, attn1_bf, w2T, outh, nullptr, nullptr, nullptr,
                           NSEQ, ND, NM, (long long)NSEQ * NM, (long long)ND * NM,
                           (long long)NSEQ * ND, 0, ND, 0, 1.f, BH);
  // 13) depthwise conv residual on v (LDS-staged, async copies when available)
  {
    dim3 cgrid(NSEQ / CT, BH, 1);
    conv_residual<<<cgrid, 256, 0, stream>>>(v_bf, wres_f, outh);
  }
  // 14) merge heads -> (B, n, F) bf16
  merge_heads<<<gblk(TOK * NF), 256, 0, stream>>>(outh, y_bf);
  // 15) out = y @ w_out^T + b_out  -> d_out (f32)
  gemm_nt<1, false, true>(stream, y_bf, wo_bf, outp, nullptr, nullptr, bout_f,
                          (int)TOK, NF, NF, 0, 0, 0, 0, NF, 0, 1.f, 1);
}
